// XYZ_TimePiecewiseConstant_81432579932517
// MI455X (gfx1250) — matrix operand order repaired and checked by reading, not compiled, not run on hardware
//
#include <hip/hip_runtime.h>
#include <stdint.h>

// ---------------------------------------------------------------------------
// XYZ_TimePiecewiseConstant for MI455X (gfx1250, wave32, WMMA)
// Fused: hash-grid embed (gather) -> 10-step GRU via v_wmma_f32_16x16x32_f16
//        -> cumsum anchors -> softmax blend -> output.
// One block = 128 points, 256 threads (8 waves), ~176KB dynamic LDS.
// ---------------------------------------------------------------------------

typedef __attribute__((ext_vector_type(16))) _Float16 v16h;
typedef __attribute__((ext_vector_type(8)))  _Float16 v8h;
typedef __attribute__((ext_vector_type(8)))  float    v8f;

static constexpr int NPIECES = 10;
static constexpr int NLEVELS = 16;
static constexpr unsigned TSIZE  = 1u << 19;
static constexpr unsigned TMASK  = TSIZE - 1u;
static constexpr int PTS  = 128;   // points per block
static constexpr int ESTR = 328;   // emb row stride in f16 (164 dwords, conflict-free-ish)
static constexpr int HSTR = 72;    // hidden row stride in f16 (36 dwords, conflict-free b128)
static constexpr int GSTR = 258;   // gates row stride in f16 (129 dwords, odd -> conflict-free)

__device__ __forceinline__ v16h ld_a16(const _Float16* p) {
  // Two ds_load_b128: elements [0..7] and [16..23] of the 32-wide K tile for
  // this lane's K-half, matching the 16-bit A-matrix VGPR layout.
  v8h lo = *(const v8h*)p;
  v8h hi = *(const v8h*)(p + 16);
  return __builtin_shufflevector(lo, hi, 0,1,2,3,4,5,6,7,8,9,10,11,12,13,14,15);
}

__device__ __forceinline__ v8f wmma_f16(v16h a, v16h b, v8f c) {
  return __builtin_amdgcn_wmma_f32_16x16x32_f16(false, a, false, b, (short)0, c, false, false);
}

__device__ __forceinline__ v16h load_brow(const float* __restrict__ row) {
  v16h f;
  const int kh = (threadIdx.x & 31) >> 4;
  const float* b = row - 16 * kh + 8 * kh;
#pragma unroll
  for (int i = 0; i < 16; ++i) f[i] = (_Float16)b[i < 8 ? i : i + 8];
  return f;
}

__device__ __forceinline__ float fast_sigmoid(float x) {
  return 1.0f / (1.0f + __expf(-x));
}
__device__ __forceinline__ float fast_tanh(float x) {
  float ax = fminf(fabsf(x), 15.0f);
  float e2 = __expf(2.0f * ax);
  float r  = (e2 - 1.0f) / (e2 + 1.0f);
  return copysignf(r, x);
}

__global__ __launch_bounds__(256) void tpc_kernel(
    const float* __restrict__ xyzt,   // (N,4)
    const float* __restrict__ tables, // (10,16,2^19,2)
    const float* __restrict__ w_ih,   // (192,32)
    const float* __restrict__ w_hh,   // (192,64)
    const float* __restrict__ w_fc,   // (1,64)
    float* __restrict__ out,          // (N,32)
    int npts)
{
  extern __shared__ char smem[];
  _Float16* embS  = (_Float16*)smem;        // PTS*ESTR f16
  _Float16* hS    = embS + PTS * ESTR;      // PTS*HSTR f16
  _Float16* gS    = hS   + PTS * HSTR;      // PTS*GSTR f16
  float*    deltaS = (float*)(gS + PTS * GSTR);   // NPIECES*PTS
  float*    wghtS  = deltaS + NPIECES * PTS;      // NPIECES*PTS
  float*    partS  = wghtS  + NPIECES * PTS;      // 2*PTS

  const int tid   = threadIdx.x;
  const int lane  = tid & 31;
  const int wave  = tid >> 5;
  const int pt    = tid & (PTS - 1);
  const int half  = tid >> 7;          // 0 or 1 (wave-uniform)

  // ---- zero hidden state ----
  {
    uint32_t* hz = (uint32_t*)hS;
#pragma unroll
    for (int i = tid; i < PTS * HSTR / 2; i += 256) hz[i] = 0u;
  }

  // ---- hash-grid embedding (each thread: 1 point x 5 pieces) ----
  const int gpt  = blockIdx.x * PTS + pt;
  const int gptc = (gpt < npts) ? gpt : (npts - 1);
  float4 P = ((const float4*)xyzt)[gptc];
  const float xc0 = fminf(fmaxf(P.x, -1.0f), 1.0f);
  const float xc1 = fminf(fmaxf(P.y, -1.0f), 1.0f);
  const float xc2 = fminf(fmaxf(P.z, -1.0f), 1.0f);

  const float res_tab[NLEVELS] = {128.f,161.f,203.f,256.f,322.f,406.f,512.f,645.f,
                                  812.f,1024.f,1290.f,1625.f,2048.f,2580.f,3250.f,4096.f};
  const float2* tab2 = (const float2*)tables;
  const int p0 = half * 5;

  for (int l = 0; l < NLEVELS; ++l) {
    const float gsz = 2.0f / res_tab[l];
    const float f0 = floorf((xc0 + 1.0f) / gsz);
    const float f1 = floorf((xc1 + 1.0f) / gsz);
    const float f2 = floorf((xc2 + 1.0f) / gsz);
    const float wx = (xc0 - (f0 * gsz - 1.0f)) / gsz;
    const float wy = (xc1 - (f1 * gsz - 1.0f)) / gsz;
    const float wz = (xc2 - (f2 * gsz - 1.0f)) / gsz;
    const uint32_t bx = (uint32_t)(int)f0;
    const uint32_t by = (uint32_t)(int)f1;
    const uint32_t bz = (uint32_t)(int)f2;
    const uint32_t hx0 = bx,               hx1 = bx + 1u;
    const uint32_t hy0 = by * 2654435761u, hy1 = (by + 1u) * 2654435761u;
    const uint32_t hz0 = bz * 805459861u,  hz1 = (bz + 1u) * 805459861u;
    uint32_t idx[8];
    idx[0] = (hx0 ^ hy0 ^ hz0) & TMASK;  idx[1] = (hx0 ^ hy0 ^ hz1) & TMASK;
    idx[2] = (hx0 ^ hy1 ^ hz0) & TMASK;  idx[3] = (hx0 ^ hy1 ^ hz1) & TMASK;
    idx[4] = (hx1 ^ hy0 ^ hz0) & TMASK;  idx[5] = (hx1 ^ hy0 ^ hz1) & TMASK;
    idx[6] = (hx1 ^ hy1 ^ hz0) & TMASK;  idx[7] = (hx1 ^ hy1 ^ hz1) & TMASK;

    float2 v[5][8];
#pragma unroll
    for (int pp = 0; pp < 5; ++pp) {
      const float2* tb = tab2 + (size_t)((p0 + pp) * NLEVELS + l) * (size_t)TSIZE;
#pragma unroll
      for (int c = 0; c < 8; ++c) v[pp][c] = tb[idx[c]];
    }
#pragma unroll
    for (int pp = 0; pp < 5; ++pp) {
      float a00 = fmaf(wx, v[pp][4].x - v[pp][0].x, v[pp][0].x);
      float a01 = fmaf(wx, v[pp][5].x - v[pp][1].x, v[pp][1].x);
      float a10 = fmaf(wx, v[pp][6].x - v[pp][2].x, v[pp][2].x);
      float a11 = fmaf(wx, v[pp][7].x - v[pp][3].x, v[pp][3].x);
      float a0  = fmaf(wy, a10 - a00, a00);
      float a1  = fmaf(wy, a11 - a01, a01);
      float fA  = fmaf(wz, a1 - a0, a0);

      float b00 = fmaf(wx, v[pp][4].y - v[pp][0].y, v[pp][0].y);
      float b01 = fmaf(wx, v[pp][5].y - v[pp][1].y, v[pp][1].y);
      float b10 = fmaf(wx, v[pp][6].y - v[pp][2].y, v[pp][2].y);
      float b11 = fmaf(wx, v[pp][7].y - v[pp][3].y, v[pp][3].y);
      float b0  = fmaf(wy, b10 - b00, b00);
      float b1  = fmaf(wy, b11 - b01, b01);
      float fB  = fmaf(wz, b1 - b0, b0);

      _Float16 hA = (_Float16)fA, hB = (_Float16)fB;
      uint32_t packed = (uint32_t)__builtin_bit_cast(uint16_t, hA)
                      | ((uint32_t)__builtin_bit_cast(uint16_t, hB) << 16);
      *(uint32_t*)&embS[pt * ESTR + (p0 + pp) * 32 + l * 2] = packed;
    }
  }

  // ---- stage B fragments in registers (ISA B-matrix layout) ----
  // Gate columns (256): [0..127] r,z combined (K=96), [128..191] i_n (K=32),
  // [192..255] h_n (K=64). Wave w owns combined tile w, plus (w<4 ? h_n tile
  // 12+w : i_n tile 8+(w-4)) -> 4-5 WMMA k-iters per wave per M-tile.
  const int col   = lane & 15;   // D/B column within tile
  const int khalf = lane >> 4;   // K-half select
  v16h B00, B01, B02, B10, B11;
  {
    int g = wave * 16 + col;                         // gate row 0..127
    B00 = load_brow(w_ih + g * 32 + khalf * 16);     // K 0..31  (emb)
    B01 = load_brow(w_hh + g * 64 + khalf * 16);     // K 32..63 (H)
    B02 = load_brow(w_hh + g * 64 + 32 + khalf * 16);// K 64..95 (H)
  }
  if (wave < 4) {
    int g = 128 + wave * 16 + col;                   // h_n rows
    B10 = load_brow(w_hh + g * 64 + khalf * 16);
    B11 = load_brow(w_hh + g * 64 + 32 + khalf * 16);
  } else {
    int g = 128 + (wave - 4) * 16 + col;             // i_n rows
    B10 = load_brow(w_ih + g * 32 + khalf * 16);
    B11 = B10; // unused
  }

  __syncthreads();

  // ---- GRU scan over 10 pieces ----
  for (int p = 0; p < NPIECES; ++p) {
    // GEMM phase: D(128x256) = [emb_p | H] (128x96) x W^T via WMMA
    for (int m = 0; m < 8; ++m) {
      const int row = m * 16 + col;
      const _Float16* eb = embS + row * ESTR + p * 32 + khalf * 8;
      const _Float16* hb = hS   + row * HSTR + khalf * 8;
      v16h A0 = ld_a16(eb);        // emb_p, K 0..31
      v16h A1 = ld_a16(hb);        // H,     K 0..31
      v16h A2 = ld_a16(hb + 32);   // H,     K 32..63

      v8f acc = {0.f,0.f,0.f,0.f,0.f,0.f,0.f,0.f};
      acc = wmma_f16(A0, B00, acc);
      acc = wmma_f16(A1, B01, acc);
      acc = wmma_f16(A2, B02, acc);
      const int rb = m * 16 + khalf * 8;
      {
        const int cc = wave * 16 + col;
#pragma unroll
        for (int i = 0; i < 8; ++i) gS[(rb + i) * GSTR + cc] = (_Float16)acc[i];
      }

      v8f acc2 = {0.f,0.f,0.f,0.f,0.f,0.f,0.f,0.f};
      if (wave < 4) {
        acc2 = wmma_f16(A1, B10, acc2);
        acc2 = wmma_f16(A2, B11, acc2);
        const int cc = (12 + wave) * 16 + col;
#pragma unroll
        for (int i = 0; i < 8; ++i) gS[(rb + i) * GSTR + cc] = (_Float16)acc2[i];
      } else {
        acc2 = wmma_f16(A0, B10, acc2);
        const int cc = (8 + (wave - 4)) * 16 + col;
#pragma unroll
        for (int i = 0; i < 8; ++i) gS[(rb + i) * GSTR + cc] = (_Float16)acc2[i];
      }
    }
    __syncthreads();

    // Gate phase: each thread handles (point pt, j in [half*32, half*32+32))
    float dpart = 0.0f;
    const int j0 = half * 32;
    for (int jj = 0; jj < 32; ++jj) {
      const int j = j0 + jj;
      float gr = (float)gS[pt * GSTR + j];
      float gz = (float)gS[pt * GSTR + 64 + j];
      float gi = (float)gS[pt * GSTR + 128 + j];
      float gh = (float)gS[pt * GSTR + 192 + j];
      float r  = fast_sigmoid(gr);
      float z  = fast_sigmoid(gz);
      float nn = fast_tanh(fmaf(r, gh, gi));
      float ho = (float)hS[pt * HSTR + j];
      float hn = fmaf(z, ho - nn, nn);          // (1-z)*n + z*h
      hS[pt * HSTR + j] = (_Float16)hn;
      dpart = fmaf(hn, w_fc[j], dpart);
    }
    partS[half * PTS + pt] = dpart;
    __syncthreads();
    if (tid < PTS)
      deltaS[p * PTS + tid] = 0.2f * (partS[tid] + partS[PTS + tid]);
  }

  // ---- anchors -> softmax weights (thread per point) ----
  if (tid < PTS) {
    float tval = xyzt[gptc * 4 + 3];
    float a = 0.0f, mx = -1e30f;
    float s[NPIECES];
#pragma unroll
    for (int p = 0; p < NPIECES; ++p) {
      a += deltaS[p * PTS + tid];
      float sv = -fabsf(tval - a) * 0.01f;
      s[p] = sv;
      mx = fmaxf(mx, sv);
    }
    float sum = 0.0f;
#pragma unroll
    for (int p = 0; p < NPIECES; ++p) { float e = __expf(s[p] - mx); s[p] = e; sum += e; }
    float inv = 1.0f / sum;
#pragma unroll
    for (int p = 0; p < NPIECES; ++p) wghtS[p * PTS + tid] = s[p] * inv;
  }
  __syncthreads();

  // ---- weighted blend -> out (thread: 16 features of one point) ----
  if (gpt < npts) {
    const int kb = half * 16;
#pragma unroll
    for (int k = 0; k < 16; ++k) {
      float accv = 0.0f;
#pragma unroll
      for (int p = 0; p < NPIECES; ++p)
        accv = fmaf(wghtS[p * PTS + pt], (float)embS[pt * ESTR + p * 32 + kb + k], accv);
      out[(size_t)gpt * 32 + kb + k] = accv;
    }
  }
}

extern "C" void kernel_launch(void* const* d_in, const int* in_sizes, int n_in,
                              void* d_out, int out_size, void* d_ws, size_t ws_size,
                              hipStream_t stream) {
  const float* xyzt = (const float*)d_in[0];
  const float* tabs = (const float*)d_in[1];
  const float* w_ih = (const float*)d_in[2];
  const float* w_hh = (const float*)d_in[3];
  const float* w_fc = (const float*)d_in[4];
  float* out = (float*)d_out;
  (void)d_ws; (void)ws_size; (void)n_in; (void)out_size;

  const int npts   = in_sizes[0] / 4;
  const int blocks = (npts + PTS - 1) / PTS;
  const size_t smem = (size_t)(PTS * ESTR + PTS * HSTR + PTS * GSTR) * 2
                    + (size_t)(2 * NPIECES * PTS + 2 * PTS) * 4;  // ~176 KB

  hipFuncSetAttribute((const void*)tpc_kernel,
                      hipFuncAttributeMaxDynamicSharedMemorySize, (int)smem);
  tpc_kernel<<<blocks, 256, smem, stream>>>(xyzt, tabs, w_ih, w_hh, w_fc, out, npts);
}
